// MorletTransform_9311489097902
// MI455X (gfx1250) — matrix operand order repaired and checked by reading, not compiled, not run on hardware
//
#include <hip/hip_runtime.h>
#include <math.h>

typedef __attribute__((ext_vector_type(16))) _Float16 v16h;
typedef __attribute__((ext_vector_type(8)))  float    v8f;

#define WINLEN   1024
#define NHARM    64
#define INV_SR   (1.0f / 16000.0f)
#define INV_TP   (1.0f / 16000.0f)
#define TWO_PI   6.28318530717958647692f

// 1/sqrt(pi * 16000)
#define NORMALIZER 0.004460312f

__global__ __launch_bounds__(128) void morlet_wmma_kernel(
    const float* __restrict__ audio,   // [frames, 1024]
    const float* __restrict__ f0,      // [frames]
    float* __restrict__ dist_out,      // [frames, 64]
    float* __restrict__ amp_out)       // [frames]
{
    __shared__ unsigned int xw[WINLEN / 2];  // windowed audio, packed f16x2
    __shared__ float trans[NHARM];
    __shared__ float s_amp;

    const int frame = blockIdx.x;
    const int tid   = threadIdx.x;
    const int wave  = tid >> 5;   // 0..3 -> harmonic group
    const int lane  = tid & 31;

    const float f0v = f0[frame];

    // ---- Stage windowed audio into LDS as packed f16 pairs (uniform loop) ----
    for (int i = tid; i < WINLEN / 2; i += 128) {
        const int n0 = 2 * i, n1 = n0 + 1;
        const float d0 = (float)(n0 - 512), d1 = (float)(n1 - 512);
        const float g0 = __expf(-d0 * d0 * INV_TP) * NORMALIZER;
        const float g1 = __expf(-d1 * d1 * INV_TP) * NORMALIZER;
        const _Float16 h0 = (_Float16)(g0 * audio[(size_t)frame * WINLEN + n0]);
        const _Float16 h1 = (_Float16)(g1 * audio[(size_t)frame * WINLEN + n1]);
        const unsigned int u =
            (unsigned int)__builtin_bit_cast(unsigned short, h0) |
            ((unsigned int)__builtin_bit_cast(unsigned short, h1) << 16);
        xw[i] = u;
    }
    __syncthreads();

    // ---- Per-lane basis setup (A row = harmonic m, ISA A-layout K offsets) ----
    const int   mA  = (wave << 4) + (lane & 15);           // harmonic row 0..63
    const float fcA = f0v * (float)(mA + 1) * INV_SR;      // turns per sample
    const int   klo = (lane < 16) ? 0 : 8;

    float cph[16], sph[16];
#pragma unroll
    for (int j = 0; j < 16; ++j) {
        // A element j holds K = klo+j (j<8) or klo+16+(j-8) = klo+8+j (j>=8)
        const int koff = (j < 8) ? (klo + j) : (klo + 8 + j);
        float t = fcA * (float)koff;
        t -= floorf(t);                                    // range-reduce in turns
        __sincosf(TWO_PI * t, &sph[j], &cph[j]);
    }

    v8f acc_re = {};
    v8f acc_im = {};

    // ---- Main loop: 32 chunks of 32 samples; 2 WMMA per chunk (Re, Im) ----
    for (int c = 0; c < WINLEN / 32; ++c) {
        float tb = fcA * (float)(c * 32);
        tb -= floorf(tb);                                  // turns -> [0,1)
        float sb, cb;
        __sincosf(TWO_PI * tb, &sb, &cb);

        union { v16h h; unsigned int u[8]; } Ac, As, Bx;
#pragma unroll
        for (int j = 0; j < 16; ++j) {
            // angle addition: cos(base+phi), sin(base+phi)
            const float cv = cb * cph[j] - sb * sph[j];
            const float sv = sb * cph[j] + cb * sph[j];
            Ac.h[j] = (_Float16)cv;
            As.h[j] = (_Float16)sv;
        }

        // B (32x16): lanes 0-15 hold K 0..7,16..23, lanes 16-31 hold K 8..15,24..31,
        // broadcast across the 16 columns -> two 4-uint LDS reads.
        const int bbase = c * 16 + ((lane < 16) ? 0 : 4);
#pragma unroll
        for (int j = 0; j < 8; ++j) Bx.u[j] = xw[bbase + ((j < 4) ? j : (j + 4))];

        acc_re = __builtin_amdgcn_wmma_f32_16x16x32_f16(
            false, Ac.h, false, Bx.h, (short)0, acc_re, false, false);
        acc_im = __builtin_amdgcn_wmma_f32_16x16x32_f16(
            false, As.h, false, Bx.h, (short)0, acc_im, false, false);
    }

    // ---- D readout: lane 0 -> M=0..7, lane 16 -> M=8..15 (per C/D layout) ----
    if (lane == 0 || lane == 16) {
        const int mbase = (wave << 4) + ((lane == 16) ? 8 : 0);
#pragma unroll
        for (int v = 0; v < 8; ++v) {
            const int   m  = mbase + v;
            const float fc = f0v * (float)(m + 1) * INV_SR;
            const float re = acc_re[v];
            const float im = acc_im[v];
            const float mag = sqrtf(re * re + im * im);
            trans[m] = (fc <= 0.5f) ? mag : 0.0f;   // Nyquist mask
        }
    }
    __syncthreads();

    if (tid == 0) {
        float s = 0.0f;
        for (int m = 0; m < NHARM; ++m) s += trans[m];
        s_amp = s;
        amp_out[frame] = fminf(fmaxf(2.0f * s, 0.0f), 1.0f);
    }
    __syncthreads();

    if (tid < NHARM) {
        dist_out[(size_t)frame * NHARM + tid] = trans[tid] / s_amp;
    }
}

extern "C" void kernel_launch(void* const* d_in, const int* in_sizes, int n_in,
                              void* d_out, int out_size, void* d_ws, size_t ws_size,
                              hipStream_t stream) {
    const float* audio = (const float*)d_in[0];   // [B,T,C,1024] f32
    const float* f0    = (const float*)d_in[1];   // [B,T,C]      f32
    const int frames   = in_sizes[1];             // B*T*C = 800

    float* dist = (float*)d_out;                              // frames*64
    float* amp  = (float*)d_out + (size_t)frames * NHARM;     // frames

    morlet_wmma_kernel<<<frames, 128, 0, stream>>>(audio, f0, dist, amp);
}
